// GATv2_4587025072293
// MI455X (gfx1250) — matrix operand order repaired and checked by reading, not compiled, not run on hardware
//
#include <hip/hip_runtime.h>
#include <hip/hip_bf16.h>

// ---------------------------------------------------------------------------
// GATv2 (2 layers, heads=1) for MI455X / gfx1250, wave32 + WMMA bf16.
// ---------------------------------------------------------------------------

typedef __attribute__((ext_vector_type(16))) __bf16 v16bf;
typedef __attribute__((ext_vector_type(8)))  float  v8f;

#define NH 128          // feature width (NFEAT == NHID == 128)

__device__ __forceinline__ unsigned short f2bf_bits(float f) {
  unsigned u = __float_as_uint(f);
  return (unsigned short)((u + 0x7fffu + ((u >> 16) & 1u)) >> 16);  // RNE
}
__device__ __forceinline__ __bf16 bits2bf(unsigned short s) {
  union { unsigned short u; __bf16 b; } c; c.u = s; return c.b;
}
// monotone float <-> uint mapping for atomicMax-based segment max
__device__ __forceinline__ unsigned fenc(float f) {
  unsigned u = __float_as_uint(f);
  return (u & 0x80000000u) ? ~u : (u | 0x80000000u);
}
__device__ __forceinline__ float fdec(unsigned u) {
  unsigned b = (u & 0x80000000u) ? (u ^ 0x80000000u) : ~u;
  return __uint_as_float(b);
}
__device__ __forceinline__ float lrelu(float x, float s) { return x > 0.0f ? x : s * x; }

// ---------------------------------------------------------------------------
// Dual GEMM: xl = X @ Wl + bl ; xr = X @ Wr + br.   X:[n,128] W:[128,128].
// One wave computes a 16-row block for all 128 output cols of both matrices.
//
// WMMA bf16 fragment layouts (ISA tables):
//  A (16x32): lane&15 = M, half = lane>>4:
//     elem i<8  -> K = ks*32 + half*8 + i
//     elem i>=8 -> K = ks*32 + 16 + half*8 + (i-8)
//  B (32x16): lane&15 = N, elem i -> K same as A (table order)
//  C/D (16x16 f32): vgpr r -> M = half*8 + r, N = lane&15
//
// Weights are staged into LDS *pre-swizzled into fragment order*:
//  fragment f = ((mat*8 + nt)*4 + ks) occupies 1 KB at offset f*1024:
//     bytes [lane*16 + i*2]        for elements i = 0..7   (chunk 0, 512 B)
//     bytes [512 + lane*16 + (i-8)*2] for elements i = 8..15 (chunk 1, 512 B)
//  so a lane fetches its whole 16-elem fragment with two ds_load_b128.
// ---------------------------------------------------------------------------
__global__ __launch_bounds__(256) void gemm_dual_kernel(
    const float* __restrict__ X,
    const float* __restrict__ Wl, const float* __restrict__ bl,
    const float* __restrict__ Wr, const float* __restrict__ br,
    float* __restrict__ xl, float* __restrict__ xr, int nrows)
{
  extern __shared__ unsigned char smem[];            // 64 KB
  unsigned short* lds16 = (unsigned short*)smem;
  const int tid = threadIdx.x;

  // Stage both weight matrices into LDS in fragment order (once per block).
  for (int idx = tid; idx < 2 * NH * NH; idx += 256) {
    const int f     = idx >> 9;         // fragment id 0..63 (512 bf16 each)
    const int pos   = idx & 511;
    const int chunk = pos >> 8;         // 0: elems 0..7, 1: elems 8..15
    const int r     = pos & 255;
    const int lane  = r >> 3;
    const int ii    = r & 7;
    const int mat   = f >> 5;
    const int nt    = (f >> 2) & 7;
    const int ks    = f & 3;
    const int k = ks * 32 + chunk * 16 + (lane >> 4) * 8 + ii;
    const int n = nt * 16 + (lane & 15);
    const float* W = mat ? Wr : Wl;
    lds16[idx] = f2bf_bits(W[k * NH + n]);
  }
  __syncthreads();

  const int wave     = tid >> 5;
  const int lane     = tid & 31;
  const int rowTile  = blockIdx.x * 8 + wave;
  const int rowTiles = (nrows + 15) >> 4;
  if (rowTile >= rowTiles) return;

  const int rowBase = rowTile * 16;
  const int m    = lane & 15;
  const int half = lane >> 4;
  const float* arow = X + (size_t)(rowBase + m) * NH;

  // Load A fragments for all 4 k-steps (reused across both weight matrices).
  v16bf afr[4];
#pragma unroll
  for (int ks = 0; ks < 4; ++ks) {
    const float* p = arow + ks * 32 + half * 8;
    float t[16];
    *(float4*)(t + 0)  = *(const float4*)(p + 0);
    *(float4*)(t + 4)  = *(const float4*)(p + 4);
    *(float4*)(t + 8)  = *(const float4*)(p + 16);
    *(float4*)(t + 12) = *(const float4*)(p + 20);
#pragma unroll
    for (int i = 0; i < 16; ++i) afr[ks][i] = bits2bf(f2bf_bits(t[i]));
  }

  const unsigned char* lanep = smem + lane * 16;

#pragma unroll
  for (int mat = 0; mat < 2; ++mat) {
    const float* bias = mat ? br : bl;
    float* out        = mat ? xr : xl;
#pragma unroll
    for (int nt = 0; nt < 8; ++nt) {
      v8f acc = {0.f, 0.f, 0.f, 0.f, 0.f, 0.f, 0.f, 0.f};
#pragma unroll
      for (int ks = 0; ks < 4; ++ks) {
        const unsigned fb = (unsigned)(((mat * 8 + nt) * 4 + ks) * 1024);
        union { uint4 q[2]; v16bf v; } u;
        u.q[0] = *(const uint4*)(lanep + fb);        // ds_load_b128 (elems 0..7)
        u.q[1] = *(const uint4*)(lanep + fb + 512);  // ds_load_b128 (elems 8..15)
        acc = __builtin_amdgcn_wmma_f32_16x16x32_bf16(
            false, afr[ks], false, u.v, (short)0, acc, false, false);
      }
      const int n = nt * 16 + (lane & 15);
      const float bv = bias[n];
#pragma unroll
      for (int r = 0; r < 8; ++r) {
        const int row = rowBase + half * 8 + r;
        out[(size_t)row * NH + n] = acc[r] + bv;
      }
    }
  }
}

// out[n][f] = bias[f];  maxenc = 0 (== -inf under fenc);  denom = 0
__global__ __launch_bounds__(256) void init_nodes_kernel(
    float* __restrict__ out, const float* __restrict__ bias,
    unsigned* __restrict__ maxenc, float* __restrict__ denom, int N)
{
  const int idx = blockIdx.x * 256 + threadIdx.x;
  if (idx < N * NH) out[idx] = bias[idx & (NH - 1)];
  if (idx < N) { maxenc[idx] = 0u; denom[idx] = 0.0f; }
}

// e[eid] = leaky_relu(xl[src]+xr[dst], 0.2) . att ;  atomicMax(maxenc[dst])
__global__ __launch_bounds__(256) void edge_scores_kernel(
    const float* __restrict__ xl, const float* __restrict__ xr,
    const float* __restrict__ att,
    const int* __restrict__ src, const int* __restrict__ dst,
    int E, int Etot, float* __restrict__ e_out, unsigned* __restrict__ maxenc)
{
  const int eid  = blockIdx.x * 8 + (threadIdx.x >> 5);
  if (eid >= Etot) return;
  const int lane = threadIdx.x & 31;
  int s, d;
  if (eid < E) { s = src[eid]; d = dst[eid]; } else { s = d = eid - E; }

  const float4 a  = *(const float4*)(att + lane * 4);
  const float4 xs = *(const float4*)(xl + (size_t)s * NH + lane * 4);
  const float4 xd = *(const float4*)(xr + (size_t)d * NH + lane * 4);
  float v = lrelu(xs.x + xd.x, 0.2f) * a.x
          + lrelu(xs.y + xd.y, 0.2f) * a.y
          + lrelu(xs.z + xd.z, 0.2f) * a.z
          + lrelu(xs.w + xd.w, 0.2f) * a.w;
#pragma unroll
  for (int off = 16; off > 0; off >>= 1) v += __shfl_xor(v, off, 32);
  if (lane == 0) {
    e_out[eid] = v;
    atomicMax(maxenc + d, fenc(v));
  }
}

// w[eid] = exp(e - max[dst]);  atomicAdd(denom[dst], w)
__global__ __launch_bounds__(256) void edge_exp_kernel(
    const float* __restrict__ e_in, const unsigned* __restrict__ maxenc,
    const int* __restrict__ dst, int E, int Etot,
    float* __restrict__ w_out, float* __restrict__ denom)
{
  const int eid = blockIdx.x * 256 + threadIdx.x;
  if (eid >= Etot) return;
  const int d = (eid < E) ? dst[eid] : eid - E;
  const float w = __expf(e_in[eid] - fdec(maxenc[d]));
  w_out[eid] = w;
  atomicAdd(denom + d, w);
}

// out[dst] += (w/denom[dst]) * xl[src]   (wave per edge, 4 feats/lane)
__global__ __launch_bounds__(256) void edge_aggr_kernel(
    const float* __restrict__ xl, const float* __restrict__ w_in,
    const float* __restrict__ denom,
    const int* __restrict__ src, const int* __restrict__ dst,
    int E, int Etot, float* __restrict__ out)
{
  const int eid  = blockIdx.x * 8 + (threadIdx.x >> 5);
  if (eid >= Etot) return;
  const int lane = threadIdx.x & 31;
  int s, d;
  if (eid < E) { s = src[eid]; d = dst[eid]; } else { s = d = eid - E; }
  const float alpha = w_in[eid] / denom[d];
  const float4 xs = *(const float4*)(xl + (size_t)s * NH + lane * 4);
  float* o = out + (size_t)d * NH + lane * 4;
  atomicAdd(o + 0, alpha * xs.x);
  atomicAdd(o + 1, alpha * xs.y);
  atomicAdd(o + 2, alpha * xs.z);
  atomicAdd(o + 3, alpha * xs.w);
}

// h = leaky_relu(h, 0.01) in place (between layers)
__global__ __launch_bounds__(256) void node_lrelu_kernel(float* __restrict__ h, int total)
{
  const int idx = blockIdx.x * 256 + threadIdx.x;
  if (idx < total) h[idx] = lrelu(h[idx], 0.01f);
}

// ---------------------------------------------------------------------------
extern "C" void kernel_launch(void* const* d_in, const int* in_sizes, int n_in,
                              void* d_out, int out_size, void* d_ws, size_t ws_size,
                              hipStream_t stream) {
  const float* x    = (const float*)d_in[0];
  const int*   eidx = (const int*)d_in[1];     // int32 on device (JAX x64 off)
  const float* Wl1  = (const float*)d_in[2];
  const float* bl1  = (const float*)d_in[3];
  const float* Wr1  = (const float*)d_in[4];
  const float* br1  = (const float*)d_in[5];
  const float* att1 = (const float*)d_in[6];
  const float* b1   = (const float*)d_in[7];
  const float* Wl2  = (const float*)d_in[8];
  const float* bl2  = (const float*)d_in[9];
  const float* Wr2  = (const float*)d_in[10];
  const float* br2  = (const float*)d_in[11];
  const float* att2 = (const float*)d_in[12];
  const float* b2   = (const float*)d_in[13];

  const int N    = in_sizes[0] / NH;
  const int E    = in_sizes[1] / 2;
  const int Etot = E + N;                      // + self loops
  const int* src = eidx;
  const int* dst = eidx + E;

  // workspace carve-out
  float*  ws     = (float*)d_ws;
  const size_t nf = (size_t)N * NH;
  float*  xl     = ws;
  float*  xr     = ws + nf;
  float*  hbuf   = ws + 2 * nf;                // layer-1 aggregate, lrelu'd in place
  float*  e_arr  = ws + 3 * nf;
  const size_t ep = (size_t)((Etot + 3) & ~3);
  float*  w_arr  = e_arr + ep;
  unsigned* mx   = (unsigned*)(w_arr + ep);
  float*  denom  = (float*)(mx + ((N + 3) & ~3));

  const int rowTiles = (N + 15) / 16;
  dim3 gGemm((rowTiles + 7) / 8);
  const size_t ldsBytes = 2 * 32 * 1024;       // 64 KB fragment-order weights
  dim3 gNodeF((N * NH + 255) / 256);
  dim3 gEdgeW((Etot + 7) / 8);
  dim3 gEdgeT((Etot + 255) / 256);
  float* out = (float*)d_out;

  // ----- layer 1 -----
  gemm_dual_kernel<<<gGemm, 256, ldsBytes, stream>>>(x, Wl1, bl1, Wr1, br1, xl, xr, N);
  init_nodes_kernel<<<gNodeF, 256, 0, stream>>>(hbuf, b1, mx, denom, N);
  edge_scores_kernel<<<gEdgeW, 256, 0, stream>>>(xl, xr, att1, src, dst, E, Etot, e_arr, mx);
  edge_exp_kernel<<<gEdgeT, 256, 0, stream>>>(e_arr, mx, dst, E, Etot, w_arr, denom);
  edge_aggr_kernel<<<gEdgeW, 256, 0, stream>>>(xl, w_arr, denom, src, dst, E, Etot, hbuf);
  node_lrelu_kernel<<<gNodeF, 256, 0, stream>>>(hbuf, N * NH);

  // ----- layer 2 -----
  gemm_dual_kernel<<<gGemm, 256, ldsBytes, stream>>>(hbuf, Wl2, bl2, Wr2, br2, xl, xr, N);
  init_nodes_kernel<<<gNodeF, 256, 0, stream>>>(out, b2, mx, denom, N);
  edge_scores_kernel<<<gEdgeW, 256, 0, stream>>>(xl, xr, att2, src, dst, E, Etot, e_arr, mx);
  edge_exp_kernel<<<gEdgeT, 256, 0, stream>>>(e_arr, mx, dst, E, Etot, w_arr, denom);
  edge_aggr_kernel<<<gEdgeW, 256, 0, stream>>>(xl, w_arr, denom, src, dst, E, Etot, out);
}
